// MultiHeadAttention_tri_10565619548561
// MI455X (gfx1250) — matrix operand order repaired and checked by reading, not compiled, not run on hardware
//
#include <hip/hip_runtime.h>

typedef __attribute__((ext_vector_type(16))) __bf16 v16bf;
typedef __attribute__((ext_vector_type(8)))  float  v8f;
typedef unsigned short ushort_t;

#define DEVINL __device__ __forceinline__

// ---------------------------------------------------------------------------
// bf16 helpers (round-to-nearest-even f32 -> bf16)
// ---------------------------------------------------------------------------
DEVINL ushort_t f2bf(float f) {
  unsigned u = __builtin_bit_cast(unsigned, f);
  unsigned r = (u + 0x7FFFu + ((u >> 16) & 1u)) >> 16;
  return (ushort_t)r;
}
DEVINL __bf16 us2bf(ushort_t s) { return __builtin_bit_cast(__bf16, s); }

DEVINL v8f wmma_bf16(v16bf a, v16bf b, v8f c) {
  // D = A(16x32) * B(32x16) + C, f32 accumulate
  return __builtin_amdgcn_wmma_f32_16x16x32_bf16(
      /*neg_a=*/false, a, /*neg_b=*/false, b,
      /*c_mod=*/(short)0, c, /*reuse_a=*/false, /*reuse_b=*/false);
}

// ---------------------------------------------------------------------------
// Fragment loaders (wave32; this code's contiguous k order, A and B alike)
// A 16x32 (MxK) bf16: lane L holds row M=L%16; kgroup=L/16;
//   VGPR p holds K = p*2 + kgroup*16  (+1 for hi half)
// rp must already point at  src + row*ld + kgroup*16  (k-offset folded in).
// ---------------------------------------------------------------------------
DEVINL v16bf load_a_at(const ushort_t* rp) {
  v16bf a;
#pragma unroll
  for (int p = 0; p < 8; ++p) {
    const int k = p * 2;
    a[2 * p]     = us2bf(rp[k]);
    a[2 * p + 1] = us2bf(rp[k + 1]);
  }
  return a;
}

DEVINL v16bf load_a_frag(const ushort_t* src, int ld, int row, int kbase, int lane) {
  const int kg = (lane >> 4) & 1;
  return load_a_at(src + (size_t)row * ld + kbase + kg * 16);
}

// B 32x16 (KxN) bf16: lane L holds column N=L%16, K = (L/16)*16 + [0..15]
// -> one contiguous 16-element (32B) load from row-major (N,K) storage.
DEVINL v16bf load_b_frag(const ushort_t* src, int ld, int n, int kbase, int lane) {
  const ushort_t* rp = src + (size_t)n * ld + kbase + ((lane >> 4) & 1) * 16;
  return *(const v16bf*)rp;
}

// ---------------------------------------------------------------------------
// Kernel 0: f32 -> bf16 conversion
// ---------------------------------------------------------------------------
__global__ void cvt_f32_to_bf16_kernel(const float* __restrict__ src,
                                       ushort_t* __restrict__ dst, int n) {
  int i = blockIdx.x * blockDim.x + threadIdx.x;
  if (i < n) dst[i] = f2bf(src[i]);
}

// ---------------------------------------------------------------------------
// Kernel 1: QKV projection with scatter. One wave = 16x64 output tile
// (one A fragment reused across 4 B fragments / 4 accumulators). Inner loop
// uses precomputed base pointers + immediate offsets only (no per-iteration
// 64-bit address math -> no spills; compiler pipelines/clauses the loads).
// C(4096 x 3072) = X(4096 x 1024) @ W^T ; column n = d*48 + kind*16 + head.
// mode 0: kinds {1,2} x d-pair  (K -> (b,h,t,d); V -> (b,h,d,t) transposed)
// mode 1: kind 0, 4 d values    (TQ -> (b,h,t,d), pre-scaled by 8)
// ---------------------------------------------------------------------------
__global__ void __launch_bounds__(256, 1)
qkv_project_kernel(const ushort_t* __restrict__ X,
                   const ushort_t* __restrict__ W,
                   ushort_t* __restrict__ Kbuf,
                   ushort_t* __restrict__ Vtbuf,
                   ushort_t* __restrict__ TQbuf,
                   int mode) {
  const int lane = threadIdx.x & 31;
  const int wave = threadIdx.x >> 5;
  const int tile = blockIdx.x * (blockDim.x >> 5) + wave;
  const int ngroups = (mode == 0) ? 32 : 16;   // groups of 4 N-tiles
  const int mt = tile / ngroups;
  const int g  = tile - mt * ngroups;

  int kind[4], dd[4];
  if (mode == 0) {
#pragma unroll
    for (int i = 0; i < 4; ++i) { kind[i] = 1 + (i & 1); dd[i] = g * 2 + (i >> 1); }
  } else {
#pragma unroll
    for (int i = 0; i < 4; ++i) { kind[i] = 0; dd[i] = g * 4 + i; }
  }

  const int m0  = mt * 16;
  const int row = m0 + (lane & 15);
  const int kg8  = ((lane >> 4) & 1) * 8;
  const int kg16 = ((lane >> 4) & 1) * 16;

  // Per-wave base pointers; inner loop addresses are base + imm(kk)
  const ushort_t* xp = X + (size_t)row * 1024 + kg16;
  const ushort_t* wp[4];
#pragma unroll
  for (int i = 0; i < 4; ++i)
    wp[i] = W + (size_t)(dd[i] * 48 + kind[i] * 16 + (lane & 15)) * 1024 + kg16;

  v8f c[4] = {};
#pragma unroll 2
  for (int kk = 0; kk < 1024; kk += 32) {
    // Speculative prefetch ~8 iterations ahead (dropped silently if OOB)
    __builtin_prefetch(xp + kk + 256, 0, 3);
    __builtin_prefetch(wp[0] + kk + 256, 0, 3);
    v16bf a = load_a_at(xp + kk);
#pragma unroll
    for (int i = 0; i < 4; ++i) {
      v16bf b = *(const v16bf*)(wp[i] + kk);
      c[i] = wmma_bf16(a, b, c[i]);
    }
  }

  // C layout: lane holds column (=head), vgpr r holds row m = r + 8*(lane/16)
  const int bb = m0 >> 11;                          // batch (t = 2048 rows)
  const int t0 = (m0 & 2047) + ((lane >> 4) & 1) * 8;
  const int hh = lane & 15;                         // head
#pragma unroll
  for (int i = 0; i < 4; ++i) {
    const int d = dd[i];
    if (kind[i] == 1) {
      ushort_t* dst = Kbuf + ((size_t)(bb * 16 + hh) * 2048) * 64 + d;
#pragma unroll
      for (int r = 0; r < 8; ++r) dst[(size_t)(t0 + r) * 64] = f2bf(c[i][r]);
    } else if (kind[i] == 2) {
      ushort_t* dst = Vtbuf + ((size_t)(bb * 16 + hh) * 64 + d) * 2048;
#pragma unroll
      for (int r = 0; r < 8; ++r) dst[t0 + r] = f2bf(c[i][r]);
    } else {
      ushort_t* dst = TQbuf + ((size_t)(bb * 16 + hh) * 2048) * 64 + d;
#pragma unroll
      for (int r = 0; r < 8; ++r) dst[(size_t)(t0 + r) * 64] = f2bf(c[i][r] * 8.0f);
    }
  }
}

// ---------------------------------------------------------------------------
// Kernel 2: flash attention. One wave = 32 query rows (2 M-tiles) of one
// (b,h): K/V B-fragments are reused across both M-tiles. V fragments are
// loaded before the softmax VALU block so their latency hides under exp/
// shuffle work. Online softmax over 32-column chunks; P re-shaped C->A
// layout through LDS; O kept in 8 f32 WMMA accumulators.
// ---------------------------------------------------------------------------
__global__ void __launch_bounds__(256, 1)
flash_attn_kernel(const ushort_t* __restrict__ TQ,
                  const ushort_t* __restrict__ Kb,
                  const ushort_t* __restrict__ Vt,
                  ushort_t* __restrict__ AO) {
  __shared__ ushort_t plds[8][2 * 16 * 32];  // two 16x32 bf16 P tiles per wave
  const int lane = threadIdx.x & 31;
  const int wave = threadIdx.x >> 5;
  const int w  = blockIdx.x * 8 + wave;      // 2048 wave jobs
  const int bh = w >> 6;                     // (b*16 + h), 64 m-groups each
  const int mg = w & 63;
  const int bb = bh >> 4;
  const int hh = bh & 15;
  const int m0 = mg * 32;
  const int kg16 = ((lane >> 4) & 1) * 16;

  const ushort_t* tq = TQ + (size_t)bh * 2048 * 64;   // (t, d) row-major
  // K base: lane's column row start (+kgroup), advanced by j0*64 per chunk
  const ushort_t* kpA = Kb + (size_t)bh * 2048 * 64 + (size_t)(lane & 15) * 64 + kg16;
  // V base: lane's d-row start (+kgroup), advanced by j0 per chunk
  const ushort_t* vpA = Vt + (size_t)bh * 64 * 2048 + (size_t)(lane & 15) * 2048 + kg16;

  v16bf a_lo[2], a_hi[2];
#pragma unroll
  for (int u = 0; u < 2; ++u) {
    const int mrow = m0 + u * 16 + (lane & 15);
    a_lo[u] = load_a_frag(tq, 64, mrow, 0, lane);   // K(=d) 0..31
    a_hi[u] = load_a_frag(tq, 64, mrow, 32, lane);  // K(=d) 32..63
  }

  float rowM[2][8], rowL[2][8];
  v8f o[2][4] = {};
#pragma unroll
  for (int u = 0; u < 2; ++u)
#pragma unroll
    for (int r = 0; r < 8; ++r) { rowM[u][r] = -1e30f; rowL[u][r] = 0.0f; }

  ushort_t* pw = &plds[wave][0];

  for (int j0 = 0; j0 < 2048; j0 += 32) {
    // Shared K fragments for this 32-key chunk
    const ushort_t* kc = kpA + (size_t)j0 * 64;
    v16bf b00 = *(const v16bf*)(kc);
    v16bf b01 = *(const v16bf*)(kc + 32);
    v16bf b10 = *(const v16bf*)(kc + 16 * 64);
    v16bf b11 = *(const v16bf*)(kc + 16 * 64 + 32);

    v8f s0[2], s1[2];
#pragma unroll
    for (int u = 0; u < 2; ++u) {
      v8f t = {};
      t = wmma_bf16(a_lo[u], b00, t);
      s0[u] = wmma_bf16(a_hi[u], b01, t);
      v8f t2 = {};
      t2 = wmma_bf16(a_lo[u], b10, t2);
      s1[u] = wmma_bf16(a_hi[u], b11, t2);
    }

    // Issue V fragment loads now; softmax VALU work below hides their latency
    v16bf vb[4];
#pragma unroll
    for (int f = 0; f < 4; ++f)
      vb[f] = *(const v16bf*)(vpA + (size_t)f * 16 * 2048 + j0);

    // Online softmax: C layout -> row r lives in vgpr r across 16 lanes
#pragma unroll
    for (int u = 0; u < 2; ++u) {
      float corr[8];
#pragma unroll
      for (int r = 0; r < 8; ++r) {
        float v = fmaxf(s0[u][r], s1[u][r]);
        v = fmaxf(v, __shfl_xor(v, 1));
        v = fmaxf(v, __shfl_xor(v, 2));
        v = fmaxf(v, __shfl_xor(v, 4));
        v = fmaxf(v, __shfl_xor(v, 8));
        const float nm = fmaxf(rowM[u][r], v);
        corr[r] = __expf(rowM[u][r] - nm);
        rowM[u][r] = nm;
        s0[u][r] = __expf(s0[u][r] - nm);
        s1[u][r] = __expf(s1[u][r] - nm);
        float rs = s0[u][r] + s1[u][r];
        rs += __shfl_xor(rs, 1);
        rs += __shfl_xor(rs, 2);
        rs += __shfl_xor(rs, 4);
        rs += __shfl_xor(rs, 8);
        rowL[u][r] = rowL[u][r] * corr[r] + rs;
      }
#pragma unroll
      for (int f = 0; f < 4; ++f)
#pragma unroll
        for (int r = 0; r < 8; ++r) o[u][f][r] *= corr[r];

      // P tile u (C layout) -> LDS
      const int mloc = ((lane >> 4) & 1) * 8;
      const int nloc = lane & 15;
      ushort_t* pu = pw + u * 512;
#pragma unroll
      for (int r = 0; r < 8; ++r) {
        pu[(mloc + r) * 32 + nloc]      = f2bf(s0[u][r]);
        pu[(mloc + r) * 32 + 16 + nloc] = f2bf(s1[u][r]);
      }
    }
    asm volatile("s_wait_dscnt 0x0" ::: "memory");
    v16bf pa[2];
#pragma unroll
    for (int u = 0; u < 2; ++u)
      pa[u] = load_a_frag(pw + u * 512, 32, lane & 15, 0, lane);
    asm volatile("s_wait_dscnt 0x0" ::: "memory");

    // O += P(16x32) @ V(32x64): V fragments shared by both M-tiles
#pragma unroll
    for (int f = 0; f < 4; ++f)
#pragma unroll
      for (int u = 0; u < 2; ++u) o[u][f] = wmma_bf16(pa[u], vb[f], o[u][f]);
  }

  // Finalize: divide by softmax denominator, write AO in (b, t, h*64 + d)
  const int dc = lane & 15;
#pragma unroll
  for (int u = 0; u < 2; ++u) {
    const int t0 = m0 + u * 16 + ((lane >> 4) & 1) * 8;
#pragma unroll
    for (int r = 0; r < 8; ++r) {
      const float inv = 1.0f / rowL[u][r];
#pragma unroll
      for (int f = 0; f < 4; ++f) {
        AO[(size_t)(bb * 2048 + t0 + r) * 1024 + hh * 64 + f * 16 + dc] =
            f2bf(o[u][f][r] * inv);
      }
    }
  }
}

// ---------------------------------------------------------------------------
// Kernel 3: output projection OUT(4096 x 1024) = AO @ Wout^T, f32 result.
// One wave = 16x64 tile, same base-pointer + immediate-offset inner loop.
// ---------------------------------------------------------------------------
__global__ void __launch_bounds__(256, 1)
out_proj_kernel(const ushort_t* __restrict__ AO,
                const ushort_t* __restrict__ W,
                float* __restrict__ OUT) {
  const int lane = threadIdx.x & 31;
  const int wave = threadIdx.x >> 5;
  const int tile = blockIdx.x * (blockDim.x >> 5) + wave;
  const int mt = tile >> 4;   // 16 groups of 4 N-tiles
  const int g  = tile & 15;
  const int m0 = mt * 16;
  const int row = m0 + (lane & 15);
  const int kg8  = ((lane >> 4) & 1) * 8;
  const int kg16 = ((lane >> 4) & 1) * 16;

  const ushort_t* xp = AO + (size_t)row * 1024 + kg16;
  const ushort_t* wp[4];
#pragma unroll
  for (int i = 0; i < 4; ++i)
    wp[i] = W + (size_t)(g * 64 + i * 16 + (lane & 15)) * 1024 + kg16;

  v8f c[4] = {};
#pragma unroll 2
  for (int kk = 0; kk < 1024; kk += 32) {
    __builtin_prefetch(xp + kk + 256, 0, 3);
    __builtin_prefetch(wp[0] + kk + 256, 0, 3);
    v16bf a = load_a_at(xp + kk);
#pragma unroll
    for (int i = 0; i < 4; ++i) {
      v16bf b = *(const v16bf*)(wp[i] + kk);
      c[i] = wmma_bf16(a, b, c[i]);
    }
  }
  const int t0 = m0 + ((lane >> 4) & 1) * 8;
#pragma unroll
  for (int i = 0; i < 4; ++i)
#pragma unroll
    for (int r = 0; r < 8; ++r)
      OUT[(size_t)(t0 + r) * 1024 + g * 64 + i * 16 + (lane & 15)] = c[i][r];
}

// ---------------------------------------------------------------------------
// Host launch
// ---------------------------------------------------------------------------
extern "C" void kernel_launch(void* const* d_in, const int* in_sizes, int n_in,
                              void* d_out, int out_size, void* d_ws, size_t ws_size,
                              hipStream_t stream) {
  (void)in_sizes; (void)n_in; (void)out_size; (void)ws_size;
  const float* x     = (const float*)d_in[0];
  const float* tx    = (const float*)d_in[1];
  const float* Wqkv  = (const float*)d_in[2];
  const float* Wtqkv = (const float*)d_in[3];
  const float* Wout  = (const float*)d_in[4];
  float* out = (float*)d_out;

  char* ws = (char*)d_ws;
  size_t off = 0;
  auto carve = [&](size_t elems) -> ushort_t* {
    ushort_t* p = (ushort_t*)(ws + off);
    off += (elems * sizeof(ushort_t) + 255) & ~(size_t)255;
    return p;
  };
  const size_t nx  = 2ull * 2048 * 1024;    // x / tx / AO elements
  const size_t nw3 = 3072ull * 1024;        // Wqkv / Wtqkv
  const size_t nwo = 1024ull * 1024;        // Wout
  const size_t nhd = 2ull * 16 * 2048 * 64; // per-head tensors

  ushort_t* xb     = carve(nx);
  ushort_t* txb    = carve(nx);
  ushort_t* wqkvb  = carve(nw3);
  ushort_t* wtqkvb = carve(nw3);
  ushort_t* woutb  = carve(nwo);
  ushort_t* Kb     = carve(nhd);   // (b,h,t,d)
  ushort_t* Vt     = carve(nhd);   // (b,h,d,t)
  ushort_t* TQ     = carve(nhd);   // (b,h,t,d), pre-scaled by 8
  ushort_t* AO     = carve(nx);    // (b,t,h*d)

  auto cvt = [&](const float* s, ushort_t* dst, size_t n) {
    cvt_f32_to_bf16_kernel<<<(int)((n + 255) / 256), 256, 0, stream>>>(s, dst, (int)n);
  };
  cvt(x, xb, nx);
  cvt(tx, txb, nx);
  cvt(Wqkv, wqkvb, nw3);
  cvt(Wtqkv, wtqkvb, nw3);
  cvt(Wout, woutb, nwo);

  // mode 0: 256 M-tiles x 32 groups; mode 1: 256 x 16 groups; 8 waves/block
  qkv_project_kernel<<<1024, 256, 0, stream>>>(xb, wqkvb, Kb, Vt, TQ, 0);
  qkv_project_kernel<<<512, 256, 0, stream>>>(txb, wtqkvb, Kb, Vt, TQ, 1);

  // 2*16*64 = 2048 wave jobs (32 query rows each), 8 waves per block
  flash_attn_kernel<<<256, 256, 0, stream>>>(TQ, Kb, Vt, AO);

  // 256 M-tiles x 16 groups
  out_proj_kernel<<<512, 256, 0, stream>>>(AO, woutb, out);
}
